// NegativeSquare_11596411699416
// MI455X (gfx1250) — matrix operand order repaired and checked by reading, not compiled, not run on hardware
//
#include <hip/hip_runtime.h>

typedef __attribute__((ext_vector_type(16))) __bf16 v16bf;
typedef __attribute__((ext_vector_type(8)))  float  v8f;

#define MROWS 16384   // B*T
#define DDIM  256
#define KCB   1024
#define BM 128
#define BN 128
#define BK 32
#define LDT 40        // ushorts per LDS tile row (80 B, 16B-aligned, bank-padded)

__device__ __forceinline__ unsigned short f32_to_bf16(float f) {
    union { float f; unsigned int u; } c; c.f = f;
    unsigned int u = c.u;
    u += 0x7FFFu + ((u >> 16) & 1u);   // round-to-nearest-even
    return (unsigned short)(u >> 16);
}

// Generic pointer to a __shared__ object: low 32 bits are the LDS byte offset
// (LDS aperture maps addr[31:0] directly; ISA §10.2).
__device__ __forceinline__ unsigned lds_off(const void* p) {
    return (unsigned)(size_t)p;
}

// Async global->LDS 16-byte copy with immediate byte offset (IOFFSET field),
// per-lane, tracked by ASYNCcnt; no VGPR staging, no per-iteration 64-bit adds.
template <int BYTE_OFF>
__device__ __forceinline__ void async_copy_b128(unsigned ldsByteOff, const void* gptr) {
    asm volatile("global_load_async_to_lds_b128 %0, %1, off offset:%2"
                 :: "v"(ldsByteOff), "v"(gptr), "n"(BYTE_OFF) : "memory");
}

__device__ __forceinline__ void wait_async0() {
    asm volatile("s_wait_asynccnt 0" ::: "memory");
}

// Compile-time loop so asm immediates can depend on the induction variable.
template <int N> struct IC { static constexpr int value = N; };
template <int I, int E> struct StaticFor {
    template <typename F>
    static __device__ __forceinline__ void run(F& f) {
        f(IC<I>{});
        StaticFor<I + 1, E>::run(f);
    }
};
template <int E> struct StaticFor<E, E> {
    template <typename F>
    static __device__ __forceinline__ void run(F&) {}
};
template <int E, typename F>
__device__ __forceinline__ void static_for(F&& f) { StaticFor<0, E>::run(f); }

// One wave32 per row: fp32 -> bf16 conversion + row sum-of-squares.
__global__ __launch_bounds__(256)
void prep_rows(const float* __restrict__ src, unsigned short* __restrict__ dstBf,
               float* __restrict__ rowSq, int nrows) {
    const int wid  = threadIdx.x >> 5;
    const int lane = threadIdx.x & 31;
    const int row  = blockIdx.x * 8 + wid;
    if (row >= nrows) return;

    const float4* s = reinterpret_cast<const float4*>(src + (size_t)row * DDIM) + lane * 2;
    float4 a = s[0], b = s[1];
    float ss = a.x*a.x + a.y*a.y + a.z*a.z + a.w*a.w
             + b.x*b.x + b.y*b.y + b.z*b.z + b.w*b.w;

    uint4 packed;
    packed.x = (unsigned)f32_to_bf16(a.x) | ((unsigned)f32_to_bf16(a.y) << 16);
    packed.y = (unsigned)f32_to_bf16(a.z) | ((unsigned)f32_to_bf16(a.w) << 16);
    packed.z = (unsigned)f32_to_bf16(b.x) | ((unsigned)f32_to_bf16(b.y) << 16);
    packed.w = (unsigned)f32_to_bf16(b.z) | ((unsigned)f32_to_bf16(b.w) << 16);
    reinterpret_cast<uint4*>(dstBf + (size_t)row * DDIM)[lane] = packed;

    for (int off = 16; off >= 1; off >>= 1) ss += __shfl_xor(ss, off, 32);
    if (lane == 0) rowSq[row] = ss;
}

// 128x128 block tile GEMM, bf16 WMMA, double-buffered LDS filled by
// GLOBAL_LOAD_ASYNC_TO_LDS_B128 (immediate k offsets), fused distance epilogue.
__global__ __launch_bounds__(256)
void vq_gemm(const unsigned short* __restrict__ Abf,
             const unsigned short* __restrict__ Cbf,
             const float* __restrict__ xSq, const float* __restrict__ cSq,
             const float* __restrict__ precision, float* __restrict__ out) {
    __shared__ unsigned short sA[2][BM * LDT];
    __shared__ unsigned short sB[2][BN * LDT];

    const int tid  = threadIdx.x;
    const int lane = tid & 31;
    const int wid  = tid >> 5;
    const int wm   = wid & 3;   // 4 waves along M
    const int wn   = wid >> 2;  // 2 waves along N
    const int m0   = blockIdx.y * BM;
    const int n0   = blockIdx.x * BN;

    const int lrow15 = lane & 15;
    const int hi     = lane >> 4;

    // Copy slots: 512 16B chunks per 128x32 tile, 2 per thread (rows r0/r1, 16B col cc).
    const int r0 = tid >> 2;
    const int r1 = r0 + 64;
    const int cc = tid & 3;
    const unsigned short* gA0 = Abf + (size_t)(m0 + r0) * DDIM + cc * 8;
    const unsigned short* gA1 = Abf + (size_t)(m0 + r1) * DDIM + cc * 8;
    const unsigned short* gB0 = Cbf + (size_t)(n0 + r0) * DDIM + cc * 8;
    const unsigned short* gB1 = Cbf + (size_t)(n0 + r1) * DDIM + cc * 8;

    unsigned lA0[2], lA1[2], lB0[2], lB1[2];
    #pragma unroll
    for (int bf = 0; bf < 2; bf++) {
        lA0[bf] = lds_off(&sA[bf][r0 * LDT + cc * 8]);
        lA1[bf] = lds_off(&sA[bf][r1 * LDT + cc * 8]);
        lB0[bf] = lds_off(&sB[bf][r0 * LDT + cc * 8]);
        lB1[bf] = lds_off(&sB[bf][r1 * LDT + cc * 8]);
    }

    v8f acc[2][4];
    v8f zero = {0.f,0.f,0.f,0.f,0.f,0.f,0.f,0.f};
    #pragma unroll
    for (int i = 0; i < 2; i++)
        #pragma unroll
        for (int j = 0; j < 4; j++) acc[i][j] = zero;

    // Prologue: fill buffer 0 with k-step 0.
    async_copy_b128<0>(lA0[0], gA0);
    async_copy_b128<0>(lA1[0], gA1);
    async_copy_b128<0>(lB0[0], gB0);
    async_copy_b128<0>(lB1[0], gB1);

    static_for<DDIM / BK>([&](auto ksc) {
        constexpr int ks  = decltype(ksc)::value;
        constexpr int cur = ks & 1;

        // Own async fills of `cur` complete; barrier also drains DScnt so the
        // other buffer's previous readers are done before we overwrite it.
        wait_async0();
        __syncthreads();

        if constexpr (ks < DDIM / BK - 1) {
            constexpr int kbyte = (ks + 1) * BK * 2;  // bytes along D
            async_copy_b128<kbyte>(lA0[cur ^ 1], gA0);
            async_copy_b128<kbyte>(lA1[cur ^ 1], gA1);
            async_copy_b128<kbyte>(lB0[cur ^ 1], gB0);
            async_copy_b128<kbyte>(lB1[cur ^ 1], gB1);
        }

        union Frag { v16bf v; uint4 q[2]; };
        Frag a[2], b[4];

        // A 16x32 bf16 frag: lanes 0-15 hold K 0-7 & 16-23; lanes 16-31 hold K 8-15 & 24-31.
        #pragma unroll
        for (int tm = 0; tm < 2; tm++) {
            int r = wm * 32 + tm * 16 + lrow15;
            const unsigned short* p = &sA[cur][r * LDT + hi * 8];
            a[tm].q[0] = *reinterpret_cast<const uint4*>(p);
            a[tm].q[1] = *reinterpret_cast<const uint4*>(p + 16);
        }
        // B 32x16 bf16 frag: lane = column; lanes 0-15 hold K 0-7 & 16-23; lanes 16-31 hold K 8-15 & 24-31.
        #pragma unroll
        for (int tn = 0; tn < 4; tn++) {
            int r = wn * 64 + tn * 16 + lrow15;
            const unsigned short* p = &sB[cur][r * LDT + hi * 8];
            b[tn].q[0] = *reinterpret_cast<const uint4*>(p);
            b[tn].q[1] = *reinterpret_cast<const uint4*>(p + 16);
        }

        #pragma unroll
        for (int tm = 0; tm < 2; tm++)
            #pragma unroll
            for (int tn = 0; tn < 4; tn++)
                acc[tm][tn] = __builtin_amdgcn_wmma_f32_16x16x32_bf16(
                    false, a[tm].v, false, b[tn].v,
                    (short)0, acc[tm][tn], false, false);
    });

    // Epilogue: out = -p*(xSq - 2*xc + cSq) = 2p*acc - p*(xSq + cSq)
    const float p  = precision[0];
    const float p2 = 2.0f * p;
    #pragma unroll
    for (int tm = 0; tm < 2; tm++) {
        #pragma unroll
        for (int tn = 0; tn < 4; tn++) {
            int colN = n0 + wn * 64 + tn * 16 + lrow15;
            float cs = cSq[colN];
            #pragma unroll
            for (int r = 0; r < 8; r++) {
                // C/D layout: VGPR r -> M=r (lanes 0-15) / M=8+r (lanes 16-31)
                int rowM = m0 + wm * 32 + tm * 16 + r + hi * 8;
                float v = p2 * acc[tm][tn][r] - p * (xSq[rowM] + cs);
                out[(size_t)rowM * KCB + colN] = v;
            }
        }
    }
}

extern "C" void kernel_launch(void* const* d_in, const int* in_sizes, int n_in,
                              void* d_out, int out_size, void* d_ws, size_t ws_size,
                              hipStream_t stream) {
    const float* x    = (const float*)d_in[0];   // [16,1024,256] f32
    const float* cb   = (const float*)d_in[1];   // [1024,256]    f32
    const float* prec = (const float*)d_in[2];   // [1]           f32
    float* out = (float*)d_out;                  // [16,1024,1024] f32

    // Workspace layout (~9 MiB): A_bf16 | C_bf16 | xSq | cSq
    unsigned short* Abf = (unsigned short*)d_ws;
    unsigned short* Cbf = Abf + (size_t)MROWS * DDIM;
    float* xSq = (float*)(Cbf + (size_t)KCB * DDIM);
    float* cSq = xSq + MROWS;

    hipLaunchKernelGGL(prep_rows, dim3(MROWS / 8), dim3(256), 0, stream, x,  Abf, xSq, MROWS);
    hipLaunchKernelGGL(prep_rows, dim3(KCB / 8),   dim3(256), 0, stream, cb, Cbf, cSq, KCB);
    hipLaunchKernelGGL(vq_gemm, dim3(KCB / BN, MROWS / BM), dim3(256), 0, stream,
                       Abf, Cbf, xSq, cSq, prec, out);
}
